// MultiHeadAttention_80728205295840
// MI455X (gfx1250) — matrix operand order repaired and checked by reading, not compiled, not run on hardware
//
#include <hip/hip_runtime.h>
#include <hip/hip_bf16.h>
#include <type_traits>
#include <cstdint>

// MHA for B=2, S=2048, D=1024, H=16, DK=64 on gfx1250 (CDNA5, wave32).
// Pipeline:
//   1. fp32->fp16 convert of q/k/v activations (one pass)
//   2. fp32->fp16 convert+transpose of Wq/Wk/Wv/Wo -> Wt[N][K]
//   3. WMMA GEMMs (per-lane async double-buffered LDS staging); V-proj
//      writes its output directly transposed (Vt[(b,h,dk)][token])
//   4. flash attention: WMMA scores + PV, DPP16 softmax reductions,
//      TDM (tensor_load_to_lds) double-buffered K/V tile staging with
//      hardware LDS padding (64 -> 72 half row stride)
//   5. out-proj WMMA GEMM -> fp32 output

typedef __attribute__((ext_vector_type(16))) _Float16     v16h;
typedef __attribute__((ext_vector_type(8)))  _Float16     v8h;
typedef __attribute__((ext_vector_type(8)))  float        v8f;
typedef __attribute__((ext_vector_type(4)))  unsigned int v4u;
typedef __attribute__((ext_vector_type(8)))  int          v8i;
typedef __attribute__((ext_vector_type(4)))  int          v4i;

#define WMMA_F16(a, b, c) \
  __builtin_amdgcn_wmma_f32_16x16x32_f16(false, (a), false, (b), (short)0, (c), false, false)

constexpr int SEQ = 2048, DMODEL = 1024, DHEAD = 64, NHEAD = 16;

// ---------------------------------------------------------------------------
// Fragment loaders (wave32, 16x16x32 f16 WMMA layouts per CDNA5 ISA 7.12.2)
// ---------------------------------------------------------------------------

// A fragment: 16(M) x 32(K); element (m,k) at p[m*ld + k0 + k].
static __device__ __forceinline__ v16h load_a_frag(const _Float16* p, int ld, int k0) {
  const int lane = threadIdx.x & 31;
  const int m  = lane & 15;
  const int kh = (lane >> 4) * 8;
  const _Float16* q = p + (size_t)m * ld + k0 + kh;
  const v8h lo = *(const v8h*)(q);        // K = kh .. kh+7
  const v8h hi = *(const v8h*)(q + 16);   // K = kh+16 .. kh+23
  return __builtin_shufflevector(lo, hi, 0,1,2,3,4,5,6,7,8,9,10,11,12,13,14,15);
}

// B fragment from K-contiguous (transposed) source: B[k][n] = p[(n0+n)*ld + k0 + k].
static __device__ __forceinline__ v16h load_b_frag_T(const _Float16* p, int ld, int n0, int k0) {
  const int lane = threadIdx.x & 31;
  const int n  = n0 + (lane & 15);
  const int kb = k0 + (lane >> 4) * 8;
  const _Float16* q = p + (size_t)n * ld + kb;
  const v8h lo = *(const v8h*)(q);
  const v8h hi = *(const v8h*)(q + 16);
  return __builtin_shufflevector(lo, hi, 0,1,2,3,4,5,6,7,8,9,10,11,12,13,14,15);
}

// ---------------------------------------------------------------------------
// DPP16 butterfly reductions across 16 lanes (xor1, xor2, xor7, xor15).
// ---------------------------------------------------------------------------
template <int CTRL>
static __device__ __forceinline__ float dpp_perm(float x) {
  const int xi = __builtin_bit_cast(int, x);
  return __builtin_bit_cast(float,
      __builtin_amdgcn_update_dpp(xi, xi, CTRL, 0xF, 0xF, true));
}
static __device__ __forceinline__ float red_max16(float x) {
  x = fmaxf(x, dpp_perm<0x0B1>(x));  // quad_perm(1,0,3,2)
  x = fmaxf(x, dpp_perm<0x04E>(x));  // quad_perm(2,3,0,1)
  x = fmaxf(x, dpp_perm<0x141>(x));  // row_half_mirror
  x = fmaxf(x, dpp_perm<0x140>(x));  // row_mirror
  return x;
}
static __device__ __forceinline__ float red_sum16(float x) {
  x += dpp_perm<0x0B1>(x);
  x += dpp_perm<0x04E>(x);
  x += dpp_perm<0x141>(x);
  x += dpp_perm<0x140>(x);
  return x;
}

// ---------------------------------------------------------------------------
// Async global -> LDS staging (ASYNCcnt-tracked; in-order completion).
// ---------------------------------------------------------------------------
static __device__ __forceinline__ void async_b128(uint32_t lds_byte_addr, const void* gaddr) {
  asm volatile("global_load_async_to_lds_b128 %0, %1, off"
               :: "v"(lds_byte_addr), "v"((unsigned long long)(uintptr_t)gaddr)
               : "memory");
}
static __device__ __forceinline__ uint32_t lds_off(const void* p) {
  return (uint32_t)(uintptr_t)p;  // low 32 bits of generic addr = LDS offset
}
static __device__ __forceinline__ void wait_async_le4() {
  asm volatile("s_wait_asynccnt 4" ::: "memory");
}
static __device__ __forceinline__ void wait_async_0() {
  asm volatile("s_wait_asynccnt 0" ::: "memory");
}

// ---------------------------------------------------------------------------
// TDM: one-instruction 2D tile DMA, global -> LDS, with hardware LDS padding.
// D# packing per CDNA5 ISA 8.3-8.4.  data_size=2B; pad 4 DWORDs (8 halves)
// after every 32 DWORDs (64 halves) => LDS row stride 72 halves.
// ---------------------------------------------------------------------------
static __device__ __forceinline__ void tdm_load_2d_64x64(
    uint32_t lds_addr, const void* gaddr, uint32_t stride_elems) {
  const unsigned long long ga = (unsigned long long)(uintptr_t)gaddr;
  v4u g0;
  g0[0] = 1u;                                        // count=1, user mode
  g0[1] = lds_addr;                                  // LDS byte address
  g0[2] = (unsigned)ga;                              // global_addr[31:0]
  g0[3] = (unsigned)(ga >> 32) | 0x80000000u;        // addr[56:32] | type=2
  v8i g1;
  g1[0] = (int)((1u << 16)                           // data_size = 2 bytes
              | (1u << 20)                           // pad_enable
              | (4u << 22)                           // pad_interval: 32 DWORDs
              | (3u << 25));                         // pad_amount: 4 DWORDs
  g1[1] = (int)(64u << 16);                          // tensor_dim0 = 64 (lo16)
  g1[2] = (int)(64u << 16);                          // dim0 hi=0 | tensor_dim1 lo16 = 64
  g1[3] = (int)(64u << 16);                          // dim1 hi=0 | tile_dim0 = 64
  g1[4] = 64;                                        // tile_dim1 = 64, tile_dim2 = 0
  g1[5] = (int)stride_elems;                         // tensor_dim0_stride lo32
  g1[6] = 0;
  g1[7] = 0;
  const v4i z = {0, 0, 0, 0};
#if __clang_major__ >= 23
  const v8i z8 = {0, 0, 0, 0, 0, 0, 0, 0};
  __builtin_amdgcn_tensor_load_to_lds(g0, g1, z, z, z8, 0);
#else
  __builtin_amdgcn_tensor_load_to_lds(g0, g1, z, z, 0);
#endif
}

// ---------------------------------------------------------------------------
// Elementwise fp32 -> fp16 (8 elems/thread, b128 in / b128 out).
// ---------------------------------------------------------------------------
__global__ __launch_bounds__(256)
void cvt_f32_f16(const float* __restrict__ x, _Float16* __restrict__ y) {
  const size_t i = ((size_t)blockIdx.x * 256 + threadIdx.x) * 8;
  const float4 a = *(const float4*)(x + i);
  const float4 b = *(const float4*)(x + i + 4);
  v8h o;
  o[0] = (_Float16)a.x; o[1] = (_Float16)a.y; o[2] = (_Float16)a.z; o[3] = (_Float16)a.w;
  o[4] = (_Float16)b.x; o[5] = (_Float16)b.y; o[6] = (_Float16)b.z; o[7] = (_Float16)b.w;
  *(v8h*)(y + i) = o;
}

// ---------------------------------------------------------------------------
// Weight convert + transpose: W[k][n] fp32 -> Wt[n][k] fp16 (64x64 LDS tiles).
// ---------------------------------------------------------------------------
__global__ __launch_bounds__(256)
void cvt_transpose_w(const float* __restrict__ W, _Float16* __restrict__ Wt) {
  __shared__ alignas(16) _Float16 t[64 * 72];
  const int n0 = blockIdx.x * 64, k0 = blockIdx.y * 64;
  const int r = threadIdx.x >> 3;         // 0..31
  const int c = (threadIdx.x & 7) * 8;    // 0..56
#pragma unroll
  for (int half = 0; half < 2; ++half) {
    const int rr = r + half * 32;
    const float4 a = *(const float4*)(W + (size_t)(k0 + rr) * DMODEL + n0 + c);
    const float4 b = *(const float4*)(W + (size_t)(k0 + rr) * DMODEL + n0 + c + 4);
    _Float16* d = &t[rr * 72 + c];
    d[0] = (_Float16)a.x; d[1] = (_Float16)a.y; d[2] = (_Float16)a.z; d[3] = (_Float16)a.w;
    d[4] = (_Float16)b.x; d[5] = (_Float16)b.y; d[6] = (_Float16)b.z; d[7] = (_Float16)b.w;
  }
  __syncthreads();
#pragma unroll
  for (int half = 0; half < 2; ++half) {
    const int rr = r + half * 32;
    v8h o;
#pragma unroll
    for (int e = 0; e < 8; ++e) o[e] = t[(c + e) * 72 + rr];
    *(v8h*)(Wt + (size_t)(n0 + rr) * DMODEL + k0 + c) = o;
  }
}

// ---------------------------------------------------------------------------
// GEMM: C = A[M,K](f16) * Wt[N,K](f16, pre-transposed) + bias.
// Block 128x128x32, 8 waves, wave tile 32x64.  Async double-buffered staging.
// TRANSO=true writes C transposed: Vt[(b*1024 + col)][token], v8h stores.
// ---------------------------------------------------------------------------
constexpr int BM = 128, BN = 128, BK = 32;
constexpr int LDS_G = BK + 8;   // 40 halves = 80B rows (16B aligned)

template <typename OT, bool TRANSO>
__global__ __launch_bounds__(256)
void gemm_wmma(const _Float16* __restrict__ A, const _Float16* __restrict__ Wt,
               const float* __restrict__ bias, OT* __restrict__ C,
               int M, int N, int K) {
  __shared__ alignas(16) _Float16 sA[2][BM * LDS_G];
  __shared__ alignas(16) _Float16 sB[2][BN * LDS_G];
  const int tid  = threadIdx.x;
  const int wid  = tid >> 5;
  const int lane = tid & 31;
  const int wm   = wid >> 1;
  const int wn   = wid & 1;
  const int row0 = blockIdx.y * BM;
  const int col0 = blockIdx.x * BN;

  const int sr = tid >> 2;            // 0..63
  const int sc = (tid & 3) * 8;       // 0,8,16,24
  const _Float16* Ab = A  + (size_t)(row0 + sr) * K + sc;
  const _Float16* Bb = Wt + (size_t)(col0 + sr) * K + sc;
  uint32_t aL[2][2], bL[2][2];
#pragma unroll
  for (int u = 0; u < 2; ++u) {
    aL[u][0] = lds_off(&sA[u][sr * LDS_G + sc]);
    aL[u][1] = lds_off(&sA[u][(sr + 64) * LDS_G + sc]);
    bL[u][0] = lds_off(&sB[u][sr * LDS_G + sc]);
    bL[u][1] = lds_off(&sB[u][(sr + 64) * LDS_G + sc]);
  }
  auto issue = [&](int buf, int k0) {
    async_b128(aL[buf][0], Ab + k0);
    async_b128(aL[buf][1], Ab + (size_t)64 * K + k0);
    async_b128(bL[buf][0], Bb + k0);
    async_b128(bL[buf][1], Bb + (size_t)64 * K + k0);
  };

  v8f acc[2][4] = {};
  const int NT = K / BK;
  issue(0, 0);
  for (int kt = 0; kt < NT; ++kt) {
    const int cur = kt & 1;
    if (kt + 1 < NT) { issue(cur ^ 1, (kt + 1) * BK); wait_async_le4(); }
    else             { wait_async_0(); }
    __syncthreads();

    const v16h a0 = load_a_frag(&sA[cur][(wm * 32 + 0)  * LDS_G], LDS_G, 0);
    const v16h a1 = load_a_frag(&sA[cur][(wm * 32 + 16) * LDS_G], LDS_G, 0);
#pragma unroll
    for (int j = 0; j < 4; ++j) {
      const v16h b = load_b_frag_T(&sB[cur][0], LDS_G, wn * 64 + j * 16, 0);
      acc[0][j] = WMMA_F16(a0, b, acc[0][j]);
      acc[1][j] = WMMA_F16(a1, b, acc[1][j]);
    }
    __syncthreads();
  }

  const int nl = lane & 15;
  const int rh = (lane >> 4) * 8;
#pragma unroll
  for (int i = 0; i < 2; ++i) {
#pragma unroll
    for (int j = 0; j < 4; ++j) {
      const int col = col0 + wn * 64 + j * 16 + nl;
      const float bb = bias[col];
      const int rbase = row0 + wm * 32 + i * 16 + rh;
      if constexpr (TRANSO) {
        // Accumulator's 8 elements are 8 consecutive tokens -> contiguous in Vt.
        const int bidx = rbase >> 11;          // token / SEQ
        const int s    = rbase & (SEQ - 1);
        v8h pk;
#pragma unroll
        for (int r = 0; r < 8; ++r) pk[r] = (_Float16)(acc[i][j][r] + bb);
        *(v8h*)((_Float16*)C + ((size_t)(bidx * DMODEL + col) * SEQ + s)) = pk;
      } else {
#pragma unroll
        for (int r = 0; r < 8; ++r) {
          C[(size_t)(rbase + r) * N + col] = (OT)(acc[i][j][r] + bb);
        }
      }
    }
  }
}

// ---------------------------------------------------------------------------
// Flash attention: workgroup = (b, h, 128-query tile); 16 rows/wave.
// 64-key steps; K and Vt tiles staged by TDM (one tensor_load_to_lds per
// tile, hardware-padded rows), double-buffered via TENSORcnt.
// 1/sqrt(dk)=0.125 is folded into the Q fragments (exact in fp16).
// ---------------------------------------------------------------------------
constexpr int LDT = 72;  // LDS row stride (halves): 64 data + 8 TDM pad

__global__ __launch_bounds__(256)
void fmha_wmma(const _Float16* __restrict__ Qm, const _Float16* __restrict__ Km,
               const _Float16* __restrict__ Vt, _Float16* __restrict__ Om) {
  __shared__ alignas(16) _Float16 sK[2][64 * LDT];   // [key j][dk]
  __shared__ alignas(16) _Float16 sV[2][64 * LDT];   // [dk][key j]
  __shared__ alignas(16) _Float16 sP[8][16 * LDT];   // wave-private P staging

  const int tid = threadIdx.x, wid = tid >> 5, lane = tid & 31;
  const int b = blockIdx.z, h = blockIdx.y;
  const int q0 = blockIdx.x * 128 + wid * 16;
  const size_t headoff = (size_t)b * SEQ * DMODEL + (size_t)h * DHEAD;

  const _Float16* qp = Qm + headoff + (size_t)q0 * DMODEL;
  v16h qa0 = load_a_frag(qp, DMODEL, 0);
  v16h qa1 = load_a_frag(qp, DMODEL, 32);
#pragma unroll
  for (int e = 0; e < 16; ++e) {        // fold softmax scale into Q (exact: 2^-3)
    qa0[e] = qa0[e] * (_Float16)0.125f;
    qa1[e] = qa1[e] * (_Float16)0.125f;
  }

  const _Float16* kp  = Km + headoff;                                // stride DMODEL
  const _Float16* vtp = Vt + (size_t)(b * NHEAD + h) * DHEAD * SEQ;  // stride SEQ

  float mrow[8], lrow[8];
#pragma unroll
  for (int r = 0; r < 8; ++r) { mrow[r] = -1e30f; lrow[r] = 0.0f; }
  v8f o[4] = {};

  const uint32_t kBase[2] = { lds_off(&sK[0][0]), lds_off(&sK[1][0]) };
  const uint32_t vBase[2] = { lds_off(&sV[0][0]), lds_off(&sV[1][0]) };
  auto issue = [&](int buf, int j0) {
    tdm_load_2d_64x64(kBase[buf], kp + (size_t)j0 * DMODEL, DMODEL);
    tdm_load_2d_64x64(vBase[buf], vtp + j0,                 SEQ);
  };

  constexpr float L2E = 1.44269504f;
  constexpr int NTJ = SEQ / 64;

  if (wid == 0) issue(0, 0);
  for (int jt = 0; jt < NTJ; ++jt) {
    const int cur = jt & 1;
    if (wid == 0) {
      if (jt + 1 < NTJ) { issue(cur ^ 1, (jt + 1) * 64); __builtin_amdgcn_s_wait_tensorcnt(2); }
      else              { __builtin_amdgcn_s_wait_tensorcnt(0); }
    }
    __syncthreads();

    // Scores: 4 key-subtiles x (DK=64 as 2 x K=32 WMMA passes).
    v8f s[4];
#pragma unroll
    for (int nn = 0; nn < 4; ++nn) {
      v8f z = {};
      z = WMMA_F16(qa0, load_b_frag_T(&sK[cur][0], LDT, nn * 16, 0),  z);
      z = WMMA_F16(qa1, load_b_frag_T(&sK[cur][0], LDT, nn * 16, 32), z);
      s[nn] = z;
    }

    // Online softmax (row's 64 scores = 4 regs across 16 lanes; DPP reduce).
    float p0[8], p1[8], p2[8], p3[8], al[8];
#pragma unroll
    for (int r = 0; r < 8; ++r) {
      const float a0 = s[0][r], a1 = s[1][r];
      const float a2 = s[2][r], a3 = s[3][r];
      const float mx = red_max16(fmaxf(fmaxf(a0, a1), fmaxf(a2, a3)));
      const float mn = fmaxf(mrow[r], mx);
      al[r] = exp2f((mrow[r] - mn) * L2E);
      p0[r] = exp2f((a0 - mn) * L2E);
      p1[r] = exp2f((a1 - mn) * L2E);
      p2[r] = exp2f((a2 - mn) * L2E);
      p3[r] = exp2f((a3 - mn) * L2E);
      const float sm = red_sum16((p0[r] + p1[r]) + (p2[r] + p3[r]));
      lrow[r] = lrow[r] * al[r] + sm;
      mrow[r] = mn;
    }
#pragma unroll
    for (int r = 0; r < 8; ++r) {
      o[0][r] *= al[r]; o[1][r] *= al[r]; o[2][r] *= al[r]; o[3][r] *= al[r];
    }

    // P (C-layout) -> wave-private LDS (row-major 16x64) -> two A fragments.
    {
      const int m = (lane >> 4) * 8;
      const int n = lane & 15;
      _Float16* pw = &sP[wid][0];
#pragma unroll
      for (int r = 0; r < 8; ++r) {
        pw[(m + r) * LDT + 0  + n] = (_Float16)p0[r];
        pw[(m + r) * LDT + 16 + n] = (_Float16)p1[r];
        pw[(m + r) * LDT + 32 + n] = (_Float16)p2[r];
        pw[(m + r) * LDT + 48 + n] = (_Float16)p3[r];
      }
    }
    asm volatile("s_wait_dscnt 0" ::: "memory");   // same-wave LDS RAW
    __builtin_amdgcn_wave_barrier();

    const v16h pa0 = load_a_frag(&sP[wid][0], LDT, 0);
    const v16h pa1 = load_a_frag(&sP[wid][0], LDT, 32);
#pragma unroll
    for (int nn = 0; nn < 4; ++nn) {
      o[nn] = WMMA_F16(pa0, load_b_frag_T(&sV[cur][0], LDT, nn * 16, 0),  o[nn]);
      o[nn] = WMMA_F16(pa1, load_b_frag_T(&sV[cur][0], LDT, nn * 16, 32), o[nn]);
    }
    __syncthreads();
  }

  // Epilogue: O /= l, store fp16 context at [b, q, h*64 + dk].
  _Float16* op = Om + headoff + (size_t)q0 * DMODEL;
  const int n  = lane & 15;
  const int rh = (lane >> 4) * 8;
#pragma unroll
  for (int r = 0; r < 8; ++r) {
    const float inv = 1.0f / lrow[r];
    op[(size_t)(rh + r) * DMODEL + 0  + n] = (_Float16)(o[0][r] * inv);
    op[(size_t)(rh + r) * DMODEL + 16 + n] = (_Float16)(o[1][r] * inv);
    op[(size_t)(rh + r) * DMODEL + 32 + n] = (_Float16)(o[2][r] * inv);
    op[(size_t)(rh + r) * DMODEL + 48 + n] = (_Float16)(o[3][r] * inv);
  }
}

// ---------------------------------------------------------------------------
// Launcher
// ---------------------------------------------------------------------------
extern "C" void kernel_launch(void* const* d_in, const int* in_sizes, int n_in,
                              void* d_out, int out_size, void* d_ws, size_t ws_size,
                              hipStream_t stream) {
  (void)in_sizes; (void)n_in; (void)out_size; (void)ws_size;
  const float* query = (const float*)d_in[0];
  const float* key   = (const float*)d_in[1];
  const float* value = (const float*)d_in[2];
  const float* Wq = (const float*)d_in[3];  const float* bq = (const float*)d_in[4];
  const float* Wk = (const float*)d_in[5];  const float* bk = (const float*)d_in[6];
  const float* Wv = (const float*)d_in[7];  const float* bv = (const float*)d_in[8];
  const float* Wo = (const float*)d_in[9];  const float* bo = (const float*)d_in[10];
  float* out = (float*)d_out;

  constexpr int M = 2 * SEQ;           // 4096 tokens
  constexpr int D = DMODEL;            // 1024
  const size_t nAct = (size_t)M * D;   // 4M halves
  const size_t nW   = (size_t)D * D;   // 1M halves

  _Float16* p = (_Float16*)d_ws;
  _Float16* xQ  = p;  p += nAct;       // fp16 activations
  _Float16* xK  = p;  p += nAct;
  _Float16* xV  = p;  p += nAct;
  _Float16* hQ  = p;  p += nAct;       // projected Q
  _Float16* hK  = p;  p += nAct;       // projected K
  _Float16* vT  = p;  p += nAct;       // projected V, transposed [(b,h,dk)][s]
  _Float16* hC  = p;  p += nAct;       // attention context
  _Float16* wQt = p;  p += nW;         // fp16 transposed weights
  _Float16* wKt = p;  p += nW;
  _Float16* wVt = p;  p += nW;
  _Float16* wOt = p;  p += nW;

  const dim3 blk(256);
  const int cvtBlocks = (int)(nAct / (256 * 8));     // 2048
  cvt_f32_f16<<<cvtBlocks, blk, 0, stream>>>(query, xQ);
  cvt_f32_f16<<<cvtBlocks, blk, 0, stream>>>(key,   xK);
  cvt_f32_f16<<<cvtBlocks, blk, 0, stream>>>(value, xV);

  const dim3 wg(D / 64, D / 64);                     // (16, 16)
  cvt_transpose_w<<<wg, blk, 0, stream>>>(Wq, wQt);
  cvt_transpose_w<<<wg, blk, 0, stream>>>(Wk, wKt);
  cvt_transpose_w<<<wg, blk, 0, stream>>>(Wv, wVt);
  cvt_transpose_w<<<wg, blk, 0, stream>>>(Wo, wOt);

  const dim3 gg(D / BN, M / BM);                     // (8, 32)
  gemm_wmma<_Float16, false><<<gg, blk, 0, stream>>>(xQ, wQt, bq, hQ, M, D, D);
  gemm_wmma<_Float16, false><<<gg, blk, 0, stream>>>(xK, wKt, bk, hK, M, D, D);
  gemm_wmma<_Float16, true ><<<gg, blk, 0, stream>>>(xV, wVt, bv, vT, M, D, D);

  fmha_wmma<<<dim3(SEQ / 128, NHEAD, 2), blk, 0, stream>>>(hQ, hK, vT, hC);

  gemm_wmma<float, false><<<gg, blk, 0, stream>>>(hC, wOt, bo, out, M, D, D);
}
